// AutoregressiveFlow_5136780886759
// MI455X (gfx1250) — matrix operand order repaired and checked by reading, not compiled, not run on hardware
//
#include <hip/hip_runtime.h>
#include <cmath>

// ---------------------------------------------------------------------------
// AutoregressiveFlow on MI455X (gfx1250, wave32, WMMA bf16 w/ f32 accumulate)
//
// One wave owns 16 batch rows. h state lives in WMMA C-layout VGPRs; each
// timestep it is restaged (bf16) into wave-private LDS and consumed as the
// WMMA A operand. Weights are staged once into LDS as bf16 with padded,
// 16B-aligned row strides. Biases ride inside the GEMM via a constant 1.0
// column in A (col 66) and a bias row in the weight buffer (k=66).
// Gates are computed in three passes (r -> n -> z/blend) to keep peak live
// accumulators ~96 VGPRs so the whole wave fits in 256 VGPRs (no
// s_set_vgpr_msb churn). Hardware V_TANH_F32 used for tanh/sigmoid.
// ---------------------------------------------------------------------------

typedef __bf16  bf16;
typedef __bf16  v16bf  __attribute__((ext_vector_type(16)));
typedef __bf16  bf16x8 __attribute__((ext_vector_type(8)));
typedef float   v8f    __attribute__((ext_vector_type(8)));

#define DEV static __device__ __forceinline__

constexpr int Bv = 32768, Tt = 64, Dd = 2, Hh = 64;
constexpr int WAVES = 8;            // waves per block
constexpr int ROWS_PER_WAVE = 16;   // WMMA M tile
// LDS strides in bf16 elements: multiples of 8 (16B-aligned rows for b128
// loads) and chosen so (stride/2) dwords spreads rows across all banks.
constexpr int SA  = 104;  // A staging: [h(0..63)|y(64,65)|1.0(66)|0(67..95)|pad]
constexpr int SW  = 104;  // combined  [W_hh | W_ih | bias | 0] rows, K=96 region
constexpr int SW1 = 72;   // W1 rows, K=64
constexpr int SW2 = 40;   // W2 rows (only the 32 needed), K=32
constexpr int SA2 = 40;   // relu-activation staging, K=32

struct __align__(16) WaveScratch {
  bf16  A[16 * SA];
  bf16  A2[16 * SA2];
  float oext[16 * 4];     // per-row {o[0], o[1], o[62], o[63]}
};
struct __align__(16) Smem {
  bf16 wcomb[192 * SW];
  bf16 w1[32 * SW1];
  bf16 w2[32 * SW2];
  WaveScratch wv[WAVES];
};

DEV v8f wmma_bf16(v16bf a, v16bf b, v8f c) {
  // v_wmma_f32_16x16x32_bf16: D = A(16x32) x B(32x16) + C, f32 accumulate
  return __builtin_amdgcn_wmma_f32_16x16x32_bf16(false, a, false, b,
                                                 (short)0, c, false, false);
}

union VU { v16bf v; bf16x8 h[2]; };

// A operand (16-bit, 16x32): lanes 0-15 hold row=lane, K runs [k0..k0+7] and
// [k0+16..k0+23]; lanes 16-31 hold the same row with K runs +8 / +24.
DEV v16bf ld_afrag(const bf16* rowbase, int k0, bool hi) {
  const int o = k0 + (hi ? 8 : 0);
  VU u;
  u.h[0] = *(const bf16x8*)(rowbase + o);
  u.h[1] = *(const bf16x8*)(rowbase + o + 16);
  return u.v;
}

// B operand (16-bit, 32x16): lane holds column n0+lane%16, K runs [k0..k0+7]
// and [k0+16..k0+23] (+8 for lanes 16-31). Buffer is row-major [n][k].
DEV v16bf ld_bfrag(const bf16* buf, int stride, int n0, int k0, int col, bool hi) {
  const bf16* p = buf + (n0 + col) * stride + k0 + (hi ? 8 : 0);
  VU u;
  u.h[0] = *(const bf16x8*)(p);
  u.h[1] = *(const bf16x8*)(p + 16);
  return u.v;
}

// CDNA5 has a hardware tanh transcendental (V_TANH_F32); TRANS ops co-execute
// with WMMA. Fall back to a 2-exp formulation if the builtin is absent.
DEV float tanh_fast(float x) {
#if __has_builtin(__builtin_amdgcn_tanhf)
  return __builtin_amdgcn_tanhf(x);
#else
  float e = __expf(2.0f * x);
  return 1.0f - 2.0f / (e + 1.0f);
#endif
}
DEV float sigmoid_fast(float x) { return 0.5f + 0.5f * tanh_fast(0.5f * x); }
DEV float softplusf_(float x) { return fmaxf(x, 0.0f) + log1pf(__expf(-fabsf(x))); }

__global__ __launch_bounds__(256) void arflow_kernel(
    const float* __restrict__ x,    const float* __restrict__ z,
    const float* __restrict__ W_ih, const float* __restrict__ W_hh,
    const float* __restrict__ b_ih, const float* __restrict__ b_hh,
    const float* __restrict__ W1,   const float* __restrict__ b1,
    const float* __restrict__ W2,   const float* __restrict__ b2,
    float* __restrict__ out)
{
  __shared__ Smem s;
  const int tid = threadIdx.x;

  // ---- stage weights to LDS (bf16, bias row at k=66, zero tails; once) ----
  for (int i = tid; i < 192 * SW; i += 256) {
    int n = i / SW, k = i % SW;
    float v;
    if (k < 64)       v = W_hh[n * 64 + k];
    else if (k < 66)  v = W_ih[n * 2 + (k - 64)];
    else if (k == 66) v = (n < 128) ? (b_ih[n] + b_hh[n]) : b_ih[n];
    else              v = 0.0f;
    s.wcomb[i] = (bf16)v;
  }
  for (int i = tid; i < 32 * SW1; i += 256) {
    int n = i / SW1, k = i % SW1;
    s.w1[i] = (bf16)((k < 64) ? W1[n * 64 + k] : 0.0f);
  }
  for (int i = tid; i < 32 * SW2; i += 256) {
    int j = i / SW2, k = i % SW2;
    int n = (j < 16) ? j : (j + 32);          // only W2 rows 0..15 and 48..63 matter
    s.w2[i] = (bf16)((k < 32) ? W2[n * 32 + k] : 0.0f);
  }
  __syncthreads();

  const int  lane   = tid & 31;
  const int  wid    = tid >> 5;
  const int  col    = lane & 15;
  const bool hi     = lane >= 16;
  const int  rowoff = hi ? 8 : 0;
  const int  base   = (blockIdx.x * WAVES + wid) * ROWS_PER_WAVE;

  bf16*  A   = s.wv[wid].A;
  bf16*  A2  = s.wv[wid].A2;
  float* ox  = s.wv[wid].oext;
  const bf16* Arow  = A  + col * SA;
  const bf16* A2row = A2 + col * SA2;

  // small bias caches (everything else rides in the GEMM via A col 66)
  float cgh[4], cb1v[2];
#pragma unroll
  for (int i = 0; i < 4; ++i) cgh[i] = b_hh[128 + i * 16 + col];
  cb1v[0] = b1[col]; cb1v[1] = b1[16 + col];
  const float cb20 = b2[col], cb23 = b2[48 + col];

  // ---- h state in C-layout: h[nt][r] = h_mat[r+rowoff][nt*16+col] ----
  v8f h[4];
#pragma unroll
  for (int nt = 0; nt < 4; ++nt)
#pragma unroll
    for (int r = 0; r < 8; ++r)
      h[nt][r] = z[(size_t)(base + r + rowoff) * Hh + nt * 16 + col];

  // initial staging: h (bf16); cols 64..95: y=0, 1.0 at col 66, zeros else
#pragma unroll
  for (int nt = 0; nt < 4; ++nt)
#pragma unroll
    for (int r = 0; r < 8; ++r)
      A[(r + rowoff) * SA + nt * 16 + col] = (bf16)h[nt][r];
  if (!hi) {
    for (int c = 64; c < 96; ++c)
      A[lane * SA + c] = (bf16)((c == 66) ? 1.0f : 0.0f);
  }

  float y0 = 0.0f, y1 = 0.0f;                 // valid in lanes 0..15 (row = lane)
  const float* xrow = x   + (size_t)(base + col) * Tt * Dd;
  float*       orow = out + (size_t)(base + col) * Tt * Dd;
  const v8f zv = {};

  for (int t = 0; t < Tt; ++t) {
    // ================= GRU: G = [h|y|1] @ [W|b].T, 3 passes =================
    // ---- pass R: n-tiles 0..3 -> r = sigmoid(...) ----
    v8f rg[4];
#pragma unroll
    for (int nt = 0; nt < 4; ++nt) rg[nt] = zv;
#pragma unroll
    for (int ks = 0; ks < 3; ++ks) {
      v16bf af = ld_afrag(Arow, ks * 32, hi);
#pragma unroll
      for (int nt = 0; nt < 4; ++nt)
        rg[nt] = wmma_bf16(af, ld_bfrag(s.wcomb, SW, nt * 16, ks * 32, col, hi), rg[nt]);
    }
#pragma unroll
    for (int i = 0; i < 4; ++i)
#pragma unroll
      for (int j = 0; j < 8; ++j) rg[i][j] = sigmoid_fast(rg[i][j]);

    // ---- pass N: gh (ksteps 0,1; bias b_hh from regs) + gi (kstep 2) ----
    v8f gh[4], gi[4];
#pragma unroll
    for (int i = 0; i < 4; ++i) {
#pragma unroll
      for (int j = 0; j < 8; ++j) gh[i][j] = cgh[i];
      gi[i] = zv;
    }
#pragma unroll
    for (int ks = 0; ks < 2; ++ks) {
      v16bf af = ld_afrag(Arow, ks * 32, hi);
#pragma unroll
      for (int i = 0; i < 4; ++i)
        gh[i] = wmma_bf16(af, ld_bfrag(s.wcomb, SW, (8 + i) * 16, ks * 32, col, hi), gh[i]);
    }
    {
      v16bf af = ld_afrag(Arow, 64, hi);      // [y | 1 | 0] columns
#pragma unroll
      for (int i = 0; i < 4; ++i)
        gi[i] = wmma_bf16(af, ld_bfrag(s.wcomb, SW, (8 + i) * 16, 64, col, hi), gi[i]);
    }
#pragma unroll
    for (int i = 0; i < 4; ++i)
#pragma unroll
      for (int j = 0; j < 8; ++j)
        gi[i][j] = tanh_fast(gi[i][j] + rg[i][j] * gh[i][j]);   // gi := n-gate

    // ---- pass Z: n-tiles 4..7 -> blend h = (1-z)*n + z*h ----
    v8f zt[4];
#pragma unroll
    for (int nt = 0; nt < 4; ++nt) zt[nt] = zv;
#pragma unroll
    for (int ks = 0; ks < 3; ++ks) {
      v16bf af = ld_afrag(Arow, ks * 32, hi);
#pragma unroll
      for (int nt = 0; nt < 4; ++nt)
        zt[nt] = wmma_bf16(af, ld_bfrag(s.wcomb, SW, (4 + nt) * 16, ks * 32, col, hi), zt[nt]);
    }
#pragma unroll
    for (int i = 0; i < 4; ++i)
#pragma unroll
      for (int j = 0; j < 8; ++j) {
        float zg = sigmoid_fast(zt[i][j]);
        h[i][j] = (1.0f - zg) * gi[i][j] + zg * h[i][j];
      }

    // restage h (wave-private LDS; same-wave DS ops are in-order -> no barrier)
#pragma unroll
    for (int nt = 0; nt < 4; ++nt)
#pragma unroll
      for (int r = 0; r < 8; ++r)
        A[(r + rowoff) * SA + nt * 16 + col] = (bf16)h[nt][r];

    // ---- MLP layer 1: a = relu(h @ W1.T + b1), 16x32, K=64 ----
    v8f am[2];
#pragma unroll
    for (int nt = 0; nt < 2; ++nt)
#pragma unroll
      for (int j = 0; j < 8; ++j) am[nt][j] = cb1v[nt];
#pragma unroll
    for (int ks = 0; ks < 2; ++ks) {
      v16bf af = ld_afrag(Arow, ks * 32, hi);
#pragma unroll
      for (int nt = 0; nt < 2; ++nt)
        am[nt] = wmma_bf16(af, ld_bfrag(s.w1, SW1, nt * 16, ks * 32, col, hi), am[nt]);
    }
#pragma unroll
    for (int nt = 0; nt < 2; ++nt)
#pragma unroll
      for (int r = 0; r < 8; ++r)
        A2[(r + rowoff) * SA2 + nt * 16 + col] = (bf16)fmaxf(am[nt][r], 0.0f);

    // ---- MLP layer 2: only o columns {0,1,62,63} are consumed ----
    v8f o0, o3;
#pragma unroll
    for (int j = 0; j < 8; ++j) { o0[j] = cb20; o3[j] = cb23; }
    {
      v16bf af = ld_afrag(A2row, 0, hi);
      o0 = wmma_bf16(af, ld_bfrag(s.w2, SW2,  0, 0, col, hi), o0);  // cols 0..15
      o3 = wmma_bf16(af, ld_bfrag(s.w2, SW2, 16, 0, col, hi), o3);  // cols 48..63
    }

    // ---- extract per-row {o0,o1,o62,o63} via wave-private LDS ----
    if (col < 2) {
#pragma unroll
      for (int r = 0; r < 8; ++r) ox[(r + rowoff) * 4 + col] = o0[r];
    }
    if (col >= 14) {
#pragma unroll
      for (int r = 0; r < 8; ++r) ox[(r + rowoff) * 4 + 2 + (col - 14)] = o3[r];
    }

    // ---- flow update: y += dloc + (softplus(s)+1e-3)*x_t (lanes 0..15) ----
    if (!hi) {
      float d0 = ox[lane * 4 + 0], d1 = ox[lane * 4 + 1];
      float s0 = ox[lane * 4 + 2], s1 = ox[lane * 4 + 3];
      float xt0 = __builtin_nontemporal_load(xrow + t * Dd + 0);
      float xt1 = __builtin_nontemporal_load(xrow + t * Dd + 1);
      y0 += d0 + (softplusf_(s0) + 0.001f) * xt0;
      y1 += d1 + (softplusf_(s1) + 0.001f) * xt1;
      __builtin_nontemporal_store(y0, orow + t * Dd + 0);
      __builtin_nontemporal_store(y1, orow + t * Dd + 1);
      A[lane * SA + 64] = (bf16)y0;   // restage y for next step's input gate
      A[lane * SA + 65] = (bf16)y1;
    }
  }
}

extern "C" void kernel_launch(void* const* d_in, const int* in_sizes, int n_in,
                              void* d_out, int out_size, void* d_ws, size_t ws_size,
                              hipStream_t stream) {
  const float* x    = (const float*)d_in[0];
  const float* z    = (const float*)d_in[1];
  const float* W_ih = (const float*)d_in[2];
  const float* W_hh = (const float*)d_in[3];
  const float* b_ih = (const float*)d_in[4];
  const float* b_hh = (const float*)d_in[5];
  const float* W1   = (const float*)d_in[6];
  const float* b1   = (const float*)d_in[7];
  const float* W2   = (const float*)d_in[8];
  const float* b2   = (const float*)d_in[9];
  float* out = (float*)d_out;

  dim3 grid(Bv / (WAVES * ROWS_PER_WAVE));   // 256 blocks x 256 threads (8 waves)
  arflow_kernel<<<grid, 256, 0, stream>>>(x, z, W_ih, W_hh, b_ih, b_hh,
                                          W1, b1, W2, b2, out);
  (void)in_sizes; (void)n_in; (void)out_size; (void)d_ws; (void)ws_size;
}
